// MultiHeadAttention_12910671691818
// MI455X (gfx1250) — matrix operand order repaired and checked by reading, not compiled, not run on hardware
//
#include <hip/hip_runtime.h>

// ---------------------------------------------------------------------------
// MHA for MI455X (gfx1250, wave32, WMMA 16x16x32 f16 with f32 accum).
// Pipeline: Q/K/V projections (f32 in -> f16 out) -> flash attention -> O proj.
// Workspace layout (f16): Q[B,H,L,64] | K[B,H,L,64] | Vt[B,H,64,L] | Ctx[B,L,D]
// = 4 * 8 MB = 32 MB.
// Data movement: attention K/Vt tiles via TDM (tensor_load_to_lds, TENSORcnt,
// with LDS pad descriptor) when available; ctx tiles via async global->LDS
// (ASYNCcnt); f32->f16 converting stages go through VGPRs.
// ---------------------------------------------------------------------------

typedef __attribute__((ext_vector_type(16))) _Float16 v16h;
typedef __attribute__((ext_vector_type(8)))  _Float16 v8h;
typedef __attribute__((ext_vector_type(4)))  _Float16 v4h;
typedef __attribute__((ext_vector_type(8)))  float    v8f;
typedef int      vi4 __attribute__((vector_size(16)));   // int4 vector
typedef int      vi8 __attribute__((vector_size(32)));   // int8 vector
typedef unsigned vu4 __attribute__((vector_size(16)));   // uint4 vector

constexpr int Bc  = 2;
constexpr int Lc  = 2048;
constexpr int Dc  = 1024;
constexpr int Hc  = 16;
constexpr int HDc = 64;
constexpr int Nc  = Bc * Lc;          // 4096 rows total

#if defined(__HIP_DEVICE_COMPILE__) &&                                   \
    __has_builtin(__builtin_amdgcn_global_load_async_to_lds_b128) &&     \
    __has_builtin(__builtin_amdgcn_s_wait_asynccnt)
#define ASYNC_LDS 1
#else
#define ASYNC_LDS 0
#endif

#if defined(__HIP_DEVICE_COMPILE__) &&                                   \
    __has_builtin(__builtin_amdgcn_tensor_load_to_lds) &&                \
    __has_builtin(__builtin_amdgcn_s_wait_tensorcnt)
#define HAVE_TDM 1
#else
#define HAVE_TDM 0
#endif

// 16-byte global -> LDS copy. Async (ASYNCcnt) when available, else via VGPRs.
__device__ __forceinline__ void cp16_g2lds(const _Float16* g, _Float16* l) {
#if ASYNC_LDS
  __builtin_amdgcn_global_load_async_to_lds_b128(
      (__attribute__((address_space(1))) vi4*)g,
      (__attribute__((address_space(3))) vi4*)l, 0, 0);
#else
  *(v8h*)l = *(const v8h*)g;
#endif
}
__device__ __forceinline__ void cp_wait() {
#if ASYNC_LDS
  __builtin_amdgcn_s_wait_asynccnt(0);  // s_wait_asynccnt 0
#endif
}

#if HAVE_TDM
// TDM: DMA a 2D f16 tile (tile_d0 x tile_d1 elems) from global (row stride
// stride0 elems) into LDS at byte offset lds_off. Pads LDS with 16B after
// every 128B row (pad_interval code 4 = 32 DWORDs, pad_amount code 3 = 4
// DWORDs) -> LDS row stride 144B = our RS=72 halves.  D# per ISA 8.3/8.4.
__device__ __forceinline__ void tdm_load_tile_f16(const _Float16* gbase,
                                                  unsigned lds_off,
                                                  unsigned tile_d0,
                                                  unsigned tile_d1,
                                                  unsigned stride0) {
  const unsigned long long ga = (unsigned long long)(size_t)gbase;
  vu4 g0;
  g0[0] = 1u;                                   // count=1, user, no gather
  g0[1] = lds_off;                              // lds_addr (bytes)
  g0[2] = (unsigned)(ga & 0xFFFFFFFFu);         // global_addr[31:0]
  g0[3] = (unsigned)((ga >> 32) & 0x01FFFFFFu)  // global_addr[56:32]
          | 0x80000000u;                        // type = 2 ("image")
  vi8 g1;
  g1[0] = (int)((1u << 16)     // data_size = 1 -> 2 bytes
                | (1u << 20)   // pad_enable
                | (4u << 22)   // pad_interval: 32 DWORDs (128B) per pad
                | (3u << 25)); // pad_amount:   4 DWORDs (16B)
  g1[1] = (int)((tile_d0 & 0xFFFFu) << 16);               // tensor_dim0 lo16
  g1[2] = (int)(((tile_d0 >> 16) & 0xFFFFu) |
                ((tile_d1 & 0xFFFFu) << 16));             // dim0 hi / dim1 lo
  g1[3] = (int)(((tile_d1 >> 16) & 0xFFFFu) |
                ((tile_d0 & 0xFFFFu) << 16));             // dim1 hi / tile_dim0
  g1[4] = (int)(tile_d1 & 0xFFFFu);                       // tile_dim1 (dim2=0)
  g1[5] = (int)stride0;                                   // dim0_stride[31:0]
  g1[6] = 0;                                              // stride hi / dim1_st
  g1[7] = 0;
  const vi4 gz  = {0, 0, 0, 0};                           // groups 2/3 unused
  const vi8 gz8 = {0, 0, 0, 0, 0, 0, 0, 0};               // 6-arg form extra
  __builtin_amdgcn_tensor_load_to_lds(g0, g1, gz, gz, gz8, 0);
}
#endif

__device__ __forceinline__ v8f wmma_f16(v16h a, v16h b, v8f c) {
  // 8 args: (neg_a, A, neg_b, B, c_mod, C, reuse_a, reuse_b)
  return __builtin_amdgcn_wmma_f32_16x16x32_f16(false, a, false, b, (short)0, c,
                                                false, false);
}

// A-fragment (16x32 f16): per lane m=lane&15, kh=lane>>4:
//   halves = [m][kh*16 .. +15]   (two 16B chunks, contiguous)
__device__ __forceinline__ v16h ldfragA(const _Float16* p) {
  v16h r;
  ((v8h*)&r)[0] = *(const v8h*)(p);
  ((v8h*)&r)[1] = *(const v8h*)(p + 8);
  return r;
}

// B-fragment (32x16 f16): per lane n=lane&15, kg=lane>>4:
//   halves = [n-th row of (W/K/Vt) tile][kg*16 .. +15]  (one 32B contiguous run)
__device__ __forceinline__ v16h ldfragB(const _Float16* p) {
  v16h r;
  ((v8h*)&r)[0] = *(const v8h*)(p);
  ((v8h*)&r)[1] = *(const v8h*)(p + 8);
  return r;
}

// ---------------------------------------------------------------------------
// Projection GEMM: out[n,d] = sum_k In[n,k] * W[d,k]   (x @ W^T)
// Block tile 128x128, BK=32. 8 waves: wave -> (wm = w>>1 in 0..3, wn = w&1),
// each wave owns a 32x64 patch = 2x4 WMMA tiles.
// MODE 0: Q out, f16 [B,H,L,64], scaled by 1/sqrt(64)=0.125
// MODE 1: K out, f16 [B,H,L,64]
// MODE 2: V out, f16 [B,H,64,L]  (transposed for PV B-fragments)
// MODE 3: final out, f32 [N,D]
// ---------------------------------------------------------------------------
template <int MODE, bool IN_F16>
__global__ __launch_bounds__(256) void proj_gemm(const void* __restrict__ Xv,
                                                 const float* __restrict__ W,
                                                 void* __restrict__ Out) {
  constexpr int BK = 32;
  constexpr int RS = 40;                    // padded LDS row stride (halves)
  __shared__ _Float16 As[128 * RS];
  __shared__ _Float16 Bs[128 * RS];

  const int t    = threadIdx.x;
  const int wave = t >> 5;
  const int lane = t & 31;
  const int rowBase = blockIdx.x * 128;
  const int colBase = blockIdx.y * 128;
  const int wm = wave >> 1;                 // 0..3 -> row offset wm*32
  const int wn = wave & 1;                  // 0..1 -> col offset wn*64
  const int mA = lane & 15;
  const int kh = lane >> 4;

  v8f acc[2][4];
#pragma unroll
  for (int mt = 0; mt < 2; ++mt)
#pragma unroll
    for (int nt = 0; nt < 4; ++nt)
#pragma unroll
      for (int i = 0; i < 8; ++i) acc[mt][nt][i] = 0.f;

  for (int k0 = 0; k0 < Dc; k0 += BK) {
    // ---- stage A tile (f16 input: async copy; f32 input: convert via VGPRs) --
    if constexpr (IN_F16) {
      const _Float16* X = (const _Float16*)Xv;
#pragma unroll
      for (int p = 0; p < 2; ++p) {
        const int r = (t >> 2) + p * 64;
        const int c = (t & 3) * 8;
        cp16_g2lds(X + (size_t)(rowBase + r) * Dc + k0 + c, As + r * RS + c);
      }
    } else {
      const float* X = (const float*)Xv;
#pragma unroll
      for (int p = 0; p < 4; ++p) {
        const int r = (t >> 3) + p * 32;
        const int c = (t & 7) * 4;
        const float4 f = *(const float4*)(X + (size_t)(rowBase + r) * Dc + k0 + c);
        v4h h;
        h[0] = (_Float16)f.x; h[1] = (_Float16)f.y;
        h[2] = (_Float16)f.z; h[3] = (_Float16)f.w;
        *(v4h*)(As + r * RS + c) = h;
      }
    }
    // ---- stage W tile (rows = output cols d, so B-frag = row read) ----
#pragma unroll
    for (int p = 0; p < 4; ++p) {
      const int r = (t >> 3) + p * 32;
      const int c = (t & 7) * 4;
      const float4 f = *(const float4*)(W + (size_t)(colBase + r) * Dc + k0 + c);
      v4h h;
      h[0] = (_Float16)f.x; h[1] = (_Float16)f.y;
      h[2] = (_Float16)f.z; h[3] = (_Float16)f.w;
      *(v4h*)(Bs + r * RS + c) = h;
    }
    if (k0 + BK < Dc) {  // hint next W tile into cache: global_prefetch_b8
      __builtin_prefetch(W + (size_t)(colBase + (t >> 3)) * Dc + k0 + BK, 0, 0);
    }
    if constexpr (IN_F16) cp_wait();
    __syncthreads();

    v16h af[2], bf[4];
#pragma unroll
    for (int mt = 0; mt < 2; ++mt)
      af[mt] = ldfragA(As + (wm * 32 + mt * 16 + mA) * RS + kh * 16);
#pragma unroll
    for (int nt = 0; nt < 4; ++nt)
      bf[nt] = ldfragB(Bs + (wn * 64 + nt * 16 + mA) * RS + kh * 16);
#pragma unroll
    for (int mt = 0; mt < 2; ++mt)
#pragma unroll
      for (int nt = 0; nt < 4; ++nt)
        acc[mt][nt] = wmma_f16(af[mt], bf[nt], acc[mt][nt]);
    __syncthreads();
  }

  // ---- epilogue: C layout is row = i + 8*(lane>>4), col = lane&15 ----
#pragma unroll
  for (int mt = 0; mt < 2; ++mt)
#pragma unroll
    for (int nt = 0; nt < 4; ++nt)
#pragma unroll
      for (int i = 0; i < 8; ++i) {
        const int n = rowBase + wm * 32 + mt * 16 + i + 8 * kh;
        const int d = colBase + wn * 64 + nt * 16 + mA;
        const float val = acc[mt][nt][i];
        if constexpr (MODE == 3) {
          ((float*)Out)[(size_t)n * Dc + d] = val;
        } else {
          const int bb = n >> 11, l = n & (Lc - 1);
          const int hh = d >> 6,  hd = d & 63;
          if constexpr (MODE == 0)
            ((_Float16*)Out)[((((size_t)bb * Hc + hh) * Lc + l) << 6) + hd] =
                (_Float16)(val * 0.125f);
          if constexpr (MODE == 1)
            ((_Float16*)Out)[((((size_t)bb * Hc + hh) * Lc + l) << 6) + hd] =
                (_Float16)val;
          if constexpr (MODE == 2)
            ((_Float16*)Out)[(((size_t)bb * Hc + hh) * HDc + hd) * Lc + l] =
                (_Float16)val;
        }
      }
}

// ---------------------------------------------------------------------------
// Flash attention: block = (b, h, 128 q rows). 8 waves x 16 q rows each.
// K tiles of 64 keys; online softmax; P routed through wave-private LDS to
// become an A-fragment for PV. Q is pre-scaled by 1/sqrt(HD).
// K/Vt tiles staged by the Tensor Data Mover (one wave issues 2 descriptors,
// TENSORcnt) when available, else async global->LDS copies (ASYNCcnt).
// ---------------------------------------------------------------------------
__global__ __launch_bounds__(256) void attn_kernel(
    const _Float16* __restrict__ Q, const _Float16* __restrict__ K,
    const _Float16* __restrict__ Vt, const unsigned char* __restrict__ mask,
    _Float16* __restrict__ Ctx) {
  constexpr int RS = 72;                     // padded row stride (halves)
  __shared__ _Float16 Ks[64 * RS];           // [lk][hd]
  __shared__ _Float16 Vs[64 * RS];           // [hd][lk]   (already transposed)
  __shared__ _Float16 Ps[8 * 16 * RS];       // per-wave P scratch [16][64]

  const int t    = threadIdx.x;
  const int wave = t >> 5;
  const int lane = t & 31;
  const int nQB  = Lc / 128;                 // 16
  const int qb   = (blockIdx.x % nQB) * 128;
  const int bh   = blockIdx.x / nQB;
  const int b    = bh / Hc, h = bh % Hc;

  const _Float16* Qbh = Q  + (size_t)bh * Lc * HDc;
  const _Float16* Kbh = K  + (size_t)bh * Lc * HDc;
  const _Float16* Vbh = Vt + (size_t)bh * HDc * Lc;
  const unsigned char* Mb = mask + (size_t)b * Lc * Lc;

  const int mA = lane & 15;
  const int kh = lane >> 4;
  const int qrow0 = qb + wave * 16;

  // Q A-fragments for this wave's 16 rows (HD=64 -> two K=32 fragments)
  v16h qf[2];
#pragma unroll
  for (int kw = 0; kw < 2; ++kw)
    qf[kw] = ldfragA(Qbh + (size_t)(qrow0 + mA) * HDc + kw * 32 + kh * 16);

  v8f ot[4];
  float rmax[8], rsum[8];
#pragma unroll
  for (int nt = 0; nt < 4; ++nt)
#pragma unroll
    for (int i = 0; i < 8; ++i) ot[nt][i] = 0.f;
#pragma unroll
  for (int i = 0; i < 8; ++i) { rmax[i] = -3.0e38f; rsum[i] = 0.f; }

  _Float16* Pw = Ps + wave * 16 * RS;

  for (int jb = 0; jb < Lc; jb += 64) {
    __syncthreads();
    // ---- stage K tile [64 x 64] and Vt tile [64 x 64] into LDS ----
#if HAVE_TDM
    if (wave == 0) {
      // generic LDS pointer: low 32 bits = LDS byte offset (flat aperture)
      tdm_load_tile_f16(Kbh + (size_t)jb * HDc,
                        (unsigned)(size_t)(void*)Ks, 64, 64, 64);
      tdm_load_tile_f16(Vbh + jb,
                        (unsigned)(size_t)(void*)Vs, 64, 64, (unsigned)Lc);
      __builtin_amdgcn_s_wait_tensorcnt(0);  // s_wait_tensorcnt 0
    }
#else
#pragma unroll
    for (int p = 0; p < 2; ++p) {
      const int r = (t >> 3) + p * 32;
      const int c = (t & 7) * 8;
      cp16_g2lds(Kbh + (size_t)(jb + r) * HDc + c, Ks + r * RS + c);
      cp16_g2lds(Vbh + (size_t)r * Lc + jb + c,    Vs + r * RS + c);
    }
    cp_wait();
#endif
    __syncthreads();

    // ---- S = Q K^T (4 N-tiles of 16 keys, K-dim = 64 = 2 WMMA) ----
    v8f st[4];
#pragma unroll
    for (int nt = 0; nt < 4; ++nt) {
      v8f z;
#pragma unroll
      for (int i = 0; i < 8; ++i) z[i] = 0.f;
      st[nt] = wmma_f16(qf[0], ldfragB(Ks + (nt * 16 + mA) * RS + kh * 16), z);
      st[nt] = wmma_f16(qf[1], ldfragB(Ks + (nt * 16 + mA) * RS + 32 + kh * 16),
                        st[nt]);
    }

    // ---- mask + online softmax ----
#pragma unroll
    for (int nt = 0; nt < 4; ++nt)
#pragma unroll
      for (int i = 0; i < 8; ++i) {
        const int lq = qrow0 + i + 8 * kh;
        const int lk = jb + nt * 16 + mA;
        if (Mb[(size_t)lq * Lc + lk]) st[nt][i] = -1.0e9f;
      }

#pragma unroll
    for (int i = 0; i < 8; ++i) {
      float mx = st[0][i];
#pragma unroll
      for (int nt = 1; nt < 4; ++nt) mx = fmaxf(mx, st[nt][i]);
#pragma unroll
      for (int off = 1; off < 16; off <<= 1)
        mx = fmaxf(mx, __shfl_xor(mx, off, 32));
      const float nm  = fmaxf(rmax[i], mx);
      const float fac = exp2f((rmax[i] - nm) * 1.4426950408889634f);
      rmax[i] = nm;
      rsum[i] *= fac;
#pragma unroll
      for (int nt = 0; nt < 4; ++nt) ot[nt][i] *= fac;
      float ps = 0.f;
#pragma unroll
      for (int nt = 0; nt < 4; ++nt) {
        const float p = exp2f((st[nt][i] - nm) * 1.4426950408889634f);
        ps += p;
        Pw[(i + 8 * kh) * RS + nt * 16 + mA] = (_Float16)p;  // wave-private
      }
      rsum[i] += ps;
    }

    // ---- O += P V  (P via LDS as A-fragments; B-frag = row of Vs) ----
    v16h pf[2];
#pragma unroll
    for (int kw = 0; kw < 2; ++kw)
      pf[kw] = ldfragA(Pw + mA * RS + kw * 32 + kh * 16);
#pragma unroll
    for (int nt = 0; nt < 4; ++nt) {
      ot[nt] = wmma_f16(pf[0], ldfragB(Vs + (nt * 16 + mA) * RS + kh * 16),
                        ot[nt]);
      ot[nt] = wmma_f16(pf[1], ldfragB(Vs + (nt * 16 + mA) * RS + 32 + kh * 16),
                        ot[nt]);
    }
  }

  // ---- finalize: cross-lane sum of rsum, normalize, store ctx (f16) ----
#pragma unroll
  for (int i = 0; i < 8; ++i) {
#pragma unroll
    for (int off = 1; off < 16; off <<= 1)
      rsum[i] += __shfl_xor(rsum[i], off, 32);
    rsum[i] = 1.0f / rsum[i];
  }
#pragma unroll
  for (int nt = 0; nt < 4; ++nt)
#pragma unroll
    for (int i = 0; i < 8; ++i) {
      const int lq = qrow0 + i + 8 * kh;
      const int dc = h * HDc + nt * 16 + mA;
      Ctx[((size_t)b * Lc + lq) * Dc + dc] = (_Float16)(ot[nt][i] * rsum[i]);
    }
}

// ---------------------------------------------------------------------------
extern "C" void kernel_launch(void* const* d_in, const int* in_sizes, int n_in,
                              void* d_out, int out_size, void* d_ws,
                              size_t ws_size, hipStream_t stream) {
  (void)in_sizes; (void)n_in; (void)out_size; (void)ws_size;
  const float* q    = (const float*)d_in[0];
  const float* k    = (const float*)d_in[1];
  const float* v    = (const float*)d_in[2];
  const float* Wq   = (const float*)d_in[3];
  const float* Wk   = (const float*)d_in[4];
  const float* Wv   = (const float*)d_in[5];
  const float* Wo   = (const float*)d_in[6];
  const unsigned char* mask = (const unsigned char*)d_in[7];  // jax bool = 1B

  const size_t MATE = (size_t)Nc * Dc;  // elements per f16 intermediate (4M)
  _Float16* Qws = (_Float16*)d_ws;
  _Float16* Kws = Qws + MATE;
  _Float16* Vws = Kws + MATE;
  _Float16* Cws = Vws + MATE;           // 32 MB total workspace

  dim3 gp(Nc / 128, Dc / 128), bp(256);
  proj_gemm<0, false><<<gp, bp, 0, stream>>>((const void*)q, Wq, (void*)Qws);
  proj_gemm<1, false><<<gp, bp, 0, stream>>>((const void*)k, Wk, (void*)Kws);
  proj_gemm<2, false><<<gp, bp, 0, stream>>>((const void*)v, Wv, (void*)Vws);

  attn_kernel<<<dim3(Bc * Hc * (Lc / 128)), dim3(256), 0, stream>>>(
      Qws, Kws, Vws, mask, Cws);

  proj_gemm<3, true><<<gp, bp, 0, stream>>>((const void*)Cws, Wo, d_out);
}
